// AttentionLayer_70179765617013
// MI455X (gfx1250) — matrix operand order repaired and checked by reading, not compiled, not run on hardware
//
#include <hip/hip_runtime.h>
#include <hip/hip_bf16.h>

// ---------------------------------------------------------------------------
// MHA layer for MI455X (gfx1250, wave32, WMMA 16x16x32 f16->f32)
//   B=8, N=1024, E=1024, H=16, D=64
// ---------------------------------------------------------------------------

typedef __attribute__((ext_vector_type(16))) _Float16 v16h;
typedef __attribute__((ext_vector_type(8)))  float    v8f;
typedef __attribute__((ext_vector_type(4)))  _Float16 v4h;
typedef __attribute__((ext_vector_type(4)))  unsigned u32x4;
typedef __attribute__((ext_vector_type(8)))  unsigned u32x8;
typedef _Float16 half_t;

union V16HCast { v16h v; uint4 q[2]; };

// A fragment: 16x32 f16 (MxK), row-major source with leading dim `ld` (halves).
__device__ __forceinline__ v16h load_frag_a(const half_t* base,
                                            int ld, int m0, int kblock, int lane) {
  const int row = m0 + (lane & 15);
  const int g   = lane >> 4;
  const half_t* p = base + (size_t)row * ld + kblock + 8 * g;
  V16HCast c;
  c.q[0] = *reinterpret_cast<const uint4*>(p);
  c.q[1] = *reinterpret_cast<const uint4*>(p + 16);
  return c.v;
}

// B fragment: 32x16 f16 (KxN), source stored [N][K] row-major (K contiguous).
__device__ __forceinline__ v16h load_frag_b(const half_t* base,
                                            int ld, int n0, int kblock, int lane) {
  const int col = n0 + (lane & 15);
  const int g   = lane >> 4;
  const half_t* p = base + (size_t)col * ld + kblock + 8 * g;
  V16HCast c;
  c.q[0] = *reinterpret_cast<const uint4*>(p);
  c.q[1] = *reinterpret_cast<const uint4*>(p + 16);
  return c.v;
}

__device__ __forceinline__ v8f wmma_f16(v16h a, v16h b, v8f c) {
  return __builtin_amdgcn_wmma_f32_16x16x32_f16(false, a, false, b,
                                                (short)0, c, false, false);
}

// ---- gfx1250 async global->LDS copy (ASYNCcnt path) -----------------------
__device__ __forceinline__ void async_copy_b128(unsigned lds_byte, const void* gaddr) {
  asm volatile("global_load_async_to_lds_b128 %0, %1, off"
               :: "v"(lds_byte), "v"(gaddr) : "memory");
}
__device__ __forceinline__ void wait_asynccnt0() {
  asm volatile("s_wait_asynccnt 0" ::: "memory");
}
__device__ __forceinline__ void wait_asynccnt8() {
  asm volatile("s_wait_asynccnt 8" ::: "memory");
}
// Truncate a generic pointer to the wave-relative LDS byte offset.
__device__ __forceinline__ unsigned lds_byte_off(const void* p) {
  return (unsigned)(size_t)p;
}

// ---- gfx1250 Tensor Data Mover: 2-D tile load, f16 elements ---------------
// Loads a tile_d1 x tile_d0 (rows x contiguous-elems) f16 tile whose row
// stride in memory is stride0 elems, into LDS with a 16-byte pad inserted
// after every 64 bytes (pad_interval=3 -> 16 DWORDs, pad_amount=3 -> 4 DWORDs)
// => LDS row stride = 80 bytes = LDS_LDA halves.
__device__ __forceinline__ void tdm_load_tile_f16(unsigned lds_byte, const void* gaddr,
                                                  unsigned tile_d0, unsigned tile_d1,
                                                  unsigned tensor_d0, unsigned tensor_d1,
                                                  unsigned stride0) {
  const unsigned long long ga = (unsigned long long)(size_t)gaddr;
  u32x4 g0;
  g0[0] = 1u;                                           // count=1 (valid D#)
  g0[1] = lds_byte;                                     // lds_addr
  g0[2] = (unsigned)ga;                                 // global_addr[31:0]
  g0[3] = (unsigned)((ga >> 32) & 0x01FFFFFFu) | (2u << 30); // addr[56:32] | type=2
  u32x8 g1;
  g1[0] = (1u << 16) | (1u << 20) | (3u << 22) | (3u << 25); // 2B, pad_en, int=3, amt=3
  g1[1] = (tensor_d0 & 0xFFFFu) << 16;                  // tensor_dim0[15:0]
  g1[2] = (tensor_d0 >> 16) | ((tensor_d1 & 0xFFFFu) << 16);
  g1[3] = (tensor_d1 >> 16) | (tile_d0 << 16);          // tile_dim0
  g1[4] = tile_d1;                                      // tile_dim1 (tile_dim2=0)
  g1[5] = stride0;                                      // tensor_dim0_stride[31:0]
  g1[6] = 0;                                            // stride0 hi | stride1 lo
  g1[7] = 0;
  asm volatile("tensor_load_to_lds %0, %1" :: "s"(g0), "s"(g1) : "memory");
}

// ---------------------------------------------------------------------------
// f32 -> f16 conversion, 4 elements/thread
// ---------------------------------------------------------------------------
__global__ void cvt_f32_f16_kernel(const float* __restrict__ src,
                                   half_t* __restrict__ dst, int n) {
  int idx = (blockIdx.x * blockDim.x + threadIdx.x) * 4;
  if (idx < n) {
    float4 f = *reinterpret_cast<const float4*>(src + idx);
    v4h o;
    o.x = (half_t)f.x; o.y = (half_t)f.y; o.z = (half_t)f.z; o.w = (half_t)f.w;
    *reinterpret_cast<v4h*>(dst + idx) = o;
  }
}

// ---------------------------------------------------------------------------
// Shared GEMM main loop: C_blk[128x128] += A[128 x 1024] * B^T (B as [N][1024]).
// Block = 128 threads (4 waves); wave tile 64x64.  Per K=32 step the A/B
// tiles (16 KB) are staged into LDS by TWO TDM descriptors issued by wave 0,
// double-buffered.  LDS leading dim = 40 halves (80B, via TDM padding) for
// bank-conflict-free b128 fragment reads.
// ---------------------------------------------------------------------------
#define LDS_LDA   40                   // halves
#define TILE_H    (128 * LDS_LDA)      // halves per A (or B) tile = 5120
#define TILE_B    (TILE_H * 2)         // bytes per tile            = 10240
#define SLOT_B    (2 * TILE_B)         // bytes per double-buffer slot (A+B)

__device__ __forceinline__ void issue_tile_tdm(unsigned lds_slot_byte,
                                               const half_t* Ag, const half_t* Bg,
                                               int browA, int bcolB, int kb,
                                               unsigned nrowsA, unsigned nrowsB) {
  tdm_load_tile_f16(lds_slot_byte,
                    Ag + (size_t)browA * 1024 + kb,
                    32u, 128u, 1024u, nrowsA, 1024u);
  tdm_load_tile_f16(lds_slot_byte + TILE_B,
                    Bg + (size_t)bcolB * 1024 + kb,
                    32u, 128u, 1024u, nrowsB, 1024u);
}

__device__ __forceinline__ void gemm_mainloop(const half_t* Ag, const half_t* Bg,
                                              int browA, int bcolB,
                                              unsigned nrowsA, unsigned nrowsB,
                                              half_t* smem,          // [2 slots]
                                              int tid, int lane,
                                              int row0w, int col0w,
                                              v8f acc[4][4]) {
  const unsigned lds0 = lds_byte_off(smem);
  const int wave = tid >> 5;

  if (wave == 0)
    issue_tile_tdm(lds0, Ag, Bg, browA, bcolB, 0, nrowsA, nrowsB);

  for (int it = 0; it < 32; ++it) {
    if (wave == 0)
      __builtin_amdgcn_s_wait_tensorcnt(0);   // this wave's TDM writes landed
    __syncthreads();

    const int cur = it & 1;
    if (wave == 0 && it + 1 < 32)
      issue_tile_tdm(lds0 + (unsigned)(cur ^ 1) * SLOT_B,
                     Ag, Bg, browA, bcolB, (it + 1) * 32, nrowsA, nrowsB);

    const half_t* As = smem + (size_t)cur * (2 * TILE_H);
    const half_t* Bs = As + TILE_H;

    v16h af[4], bf[4];
    #pragma unroll
    for (int i = 0; i < 4; ++i) af[i] = load_frag_a(As, LDS_LDA, row0w + 16 * i, 0, lane);
    #pragma unroll
    for (int j = 0; j < 4; ++j) bf[j] = load_frag_b(Bs, LDS_LDA, col0w + 16 * j, 0, lane);
    #pragma unroll
    for (int i = 0; i < 4; ++i)
      #pragma unroll
      for (int j = 0; j < 4; ++j)
        acc[i][j] = wmma_f16(af[i], bf[j], acc[i][j]);
  }
}

// ---------------------------------------------------------------------------
// QKV GEMM: C[8192 x 3072] = Xh * Wh^T (+bias) -> q[b,h,n,d], k[b,h,n,d],
// v^T[b,h,d,n] in f16.
// ---------------------------------------------------------------------------
__global__ __launch_bounds__(128)
void qkv_gemm_kernel(const half_t* __restrict__ xh, const half_t* __restrict__ wh,
                     const float* __restrict__ bias,
                     half_t* __restrict__ qh, half_t* __restrict__ kh,
                     half_t* __restrict__ vth) {
  __shared__ __align__(16) half_t smem[2][2 * TILE_H];

  const int tid  = threadIdx.x;
  const int lane = tid & 31;
  const int wave = tid >> 5;
  const int browA = blockIdx.y * 128;
  const int bcolB = blockIdx.x * 128;
  const int row0w = (wave >> 1) * 64;
  const int col0w = (wave & 1) * 64;

  v8f zero = {};
  v8f acc[4][4];
  #pragma unroll
  for (int i = 0; i < 4; ++i)
    #pragma unroll
    for (int j = 0; j < 4; ++j) acc[i][j] = zero;

  gemm_mainloop(xh, wh, browA, bcolB, 8192u, 3072u,
                &smem[0][0], tid, lane, row0w, col0w, acc);

  const int g = lane >> 4, ln = lane & 15;
  const int row0 = browA + row0w, col0 = bcolB + col0w;
  #pragma unroll
  for (int j = 0; j < 4; ++j) {
    const int n = col0 + 16 * j + ln;
    const float bj = bias[n];
    const int t = n >> 10;            // 0=q 1=k 2=v
    const int e = n & 1023;
    const int h = e >> 6, d = e & 63;
    #pragma unroll
    for (int i = 0; i < 4; ++i) {
      #pragma unroll
      for (int r = 0; r < 8; ++r) {
        const int m  = row0 + 16 * i + r + 8 * g;
        const int b  = m >> 10, nn = m & 1023;
        const int bh = b * 16 + h;
        const half_t hv = (half_t)(acc[i][j][r] + bj);
        if (t == 0)      qh[((size_t)bh * 1024 + nn) * 64 + d]  = hv;
        else if (t == 1) kh[((size_t)bh * 1024 + nn) * 64 + d]  = hv;
        else             vth[((size_t)bh * 64 + d) * 1024 + nn] = hv;
      }
    }
  }
}

// ---------------------------------------------------------------------------
// Output projection: out[8192 x 1024] = attnh * Wout^T + bias (f32 out)
// ---------------------------------------------------------------------------
__global__ __launch_bounds__(128)
void out_gemm_kernel(const half_t* __restrict__ ah, const half_t* __restrict__ wh,
                     const float* __restrict__ bias, float* __restrict__ out) {
  __shared__ __align__(16) half_t smem[2][2 * TILE_H];

  const int tid  = threadIdx.x;
  const int lane = tid & 31;
  const int wave = tid >> 5;
  const int browA = blockIdx.y * 128;
  const int bcolB = blockIdx.x * 128;
  const int row0w = (wave >> 1) * 64;
  const int col0w = (wave & 1) * 64;

  v8f zero = {};
  v8f acc[4][4];
  #pragma unroll
  for (int i = 0; i < 4; ++i)
    #pragma unroll
    for (int j = 0; j < 4; ++j) acc[i][j] = zero;

  gemm_mainloop(ah, wh, browA, bcolB, 8192u, 1024u,
                &smem[0][0], tid, lane, row0w, col0w, acc);

  const int g = lane >> 4, ln = lane & 15;
  const int row0 = browA + row0w, col0 = bcolB + col0w;
  #pragma unroll
  for (int j = 0; j < 4; ++j) {
    const int n = col0 + 16 * j + ln;
    const float bj = bias[n];
    #pragma unroll
    for (int i = 0; i < 4; ++i) {
      #pragma unroll
      for (int r = 0; r < 8; ++r) {
        const int m = row0 + 16 * i + r + 8 * g;
        out[(size_t)m * 1024 + n] = acc[i][j][r] + bj;
      }
    }
  }
}

// ---------------------------------------------------------------------------
// Flash attention: one wave per (b, h, 16-query tile). 8192 blocks of 32 thr.
// K-tiles (32x64 f16, padded to 144B rows) are double-buffered into LDS with
// async global->LDS copies (ASYNCcnt); V read direct + prefetch.
// ---------------------------------------------------------------------------
#define KLD   72                        // halves per LDS K-tile row (144 B)
#define KTH   (32 * KLD)                // halves per K tile buffer

__device__ __forceinline__ void issue_k_copy(unsigned kbase_byte,
                                             const half_t* Kb, int kb, int lane) {
  // 32 rows x 128B; lane copies 8 x 16B chunks: row = 4c + (lane>>3), seg = lane&7
  const int rl = lane >> 3, seg = lane & 7;
  #pragma unroll
  for (int c = 0; c < 4; ++c) {
    const int row = 4 * c + rl;                 // c covers rows 0..15
    async_copy_b128(kbase_byte + (unsigned)row * 144 + seg * 16,
                    Kb + (size_t)(kb + row) * 64 + seg * 8);
    const int row2 = 16 + 4 * c + rl;           // rows 16..31
    async_copy_b128(kbase_byte + (unsigned)row2 * 144 + seg * 16,
                    Kb + (size_t)(kb + row2) * 64 + seg * 8);
  }
}

__global__ __launch_bounds__(32)
void attn_kernel(const half_t* __restrict__ qh, const half_t* __restrict__ kh,
                 const half_t* __restrict__ vth, half_t* __restrict__ attnh) {
  __shared__ __align__(16) half_t kbuf[2][KTH];
  __shared__ __align__(16) half_t pbuf[16 * 32];

  const int lane = threadIdx.x & 31;
  const int tile = blockIdx.x;
  const int qt = tile & 63;        // query tile within sequence
  const int bh = tile >> 6;        // b*16 + h
  const int b  = bh >> 4, h = bh & 15;

  const half_t* Qb = qh + ((size_t)bh * 1024 + qt * 16) * 64; // 16x64, ld 64
  const half_t* Kb = kh + (size_t)bh * 1024 * 64;             // [1024][64]
  const half_t* Vt = vth + (size_t)bh * 64 * 1024;            // [64][1024]
  const unsigned kbuf0 = lds_byte_off(&kbuf[0][0]);

  const v16h qa0 = load_frag_a(Qb, 64, 0, 0, lane);
  const v16h qa1 = load_frag_a(Qb, 64, 0, 32, lane);

  v8f zero = {};
  v8f o[4] = {zero, zero, zero, zero};
  float mx[8], ls[8];
  #pragma unroll
  for (int r = 0; r < 8; ++r) { mx[r] = -1e30f; ls[r] = 0.0f; }

  const float scale = 0.125f;  // 1/sqrt(64)
  const int g = lane >> 4, ln = lane & 15;

  issue_k_copy(kbuf0, Kb, 0, lane);

  for (int kb = 0; kb < 1024; kb += 32) {
    const int cur = (kb >> 5) & 1;
    if (kb + 32 < 1024) {
      issue_k_copy(kbuf0 + (unsigned)(cur ^ 1) * (KTH * 2), Kb, kb + 32, lane);
      wait_asynccnt8();            // current tile's 8 copies completed
      // prefetch next V block (global_prefetch_b8)
      __builtin_prefetch(Vt + (size_t)(2 * lane) * 1024 + kb + 32, 0, 3);
      __builtin_prefetch(Vt + (size_t)(2 * lane + 1) * 1024 + kb + 32, 0, 3);
    } else {
      wait_asynccnt0();
    }

    const half_t* Ks = &kbuf[cur][0];
    v16h b00 = load_frag_b(Ks, KLD, 0,  0,  lane);
    v16h b01 = load_frag_b(Ks, KLD, 0,  32, lane);
    v16h b10 = load_frag_b(Ks, KLD, 16, 0,  lane);
    v16h b11 = load_frag_b(Ks, KLD, 16, 32, lane);

    v8f s0 = zero, s1 = zero;
    s0 = wmma_f16(qa0, b00, s0);
    s0 = wmma_f16(qa1, b01, s0);
    s1 = wmma_f16(qa0, b10, s1);
    s1 = wmma_f16(qa1, b11, s1);

    #pragma unroll
    for (int r = 0; r < 8; ++r) {
      float a = s0[r] * scale;
      float c = s1[r] * scale;
      float m = fmaxf(a, c);
      m = fmaxf(m, __shfl_xor(m, 1, 32));
      m = fmaxf(m, __shfl_xor(m, 2, 32));
      m = fmaxf(m, __shfl_xor(m, 4, 32));
      m = fmaxf(m, __shfl_xor(m, 8, 32));

      const float mnew  = fmaxf(mx[r], m);
      const float alpha = __expf(mx[r] - mnew);
      const float p0 = __expf(a - mnew);
      const float p1 = __expf(c - mnew);
      float rs = p0 + p1;
      rs += __shfl_xor(rs, 1, 32);
      rs += __shfl_xor(rs, 2, 32);
      rs += __shfl_xor(rs, 4, 32);
      rs += __shfl_xor(rs, 8, 32);

      ls[r] = ls[r] * alpha + rs;
      mx[r] = mnew;
      #pragma unroll
      for (int nf = 0; nf < 4; ++nf) o[nf][r] = o[nf][r] * alpha;

      const int mrow = r + 8 * g;
      pbuf[mrow * 32 + ln]      = (half_t)p0;
      pbuf[mrow * 32 + 16 + ln] = (half_t)p1;
    }
    __syncthreads();

    const v16h pa = load_frag_a(pbuf, 32, 0, 0, lane);
    #pragma unroll
    for (int nf = 0; nf < 4; ++nf) {
      v16h vb = load_frag_b(Vt, 1024, nf * 16, kb, lane);
      o[nf] = wmma_f16(pa, vb, o[nf]);
    }
    __syncthreads();
  }

  // normalize and store into attn[b][n][h*64+d] (f16, row-major E=1024)
  half_t* Orow = attnh + ((size_t)b * 1024 + qt * 16) * 1024 + h * 64;
  #pragma unroll
  for (int nf = 0; nf < 4; ++nf) {
    #pragma unroll
    for (int r = 0; r < 8; ++r) {
      const int m = r + 8 * g;
      const int d = nf * 16 + ln;
      Orow[(size_t)m * 1024 + d] = (half_t)(o[nf][r] / ls[r]);
    }
  }
}

// ---------------------------------------------------------------------------
extern "C" void kernel_launch(void* const* d_in, const int* in_sizes, int n_in,
                              void* d_out, int out_size, void* d_ws, size_t ws_size,
                              hipStream_t stream) {
  const float* x      = (const float*)d_in[0];  // [8,1024,1024]
  const float* qkv_w  = (const float*)d_in[1];  // [3072,1024]
  const float* qkv_b  = (const float*)d_in[2];  // [3072]
  const float* out_w  = (const float*)d_in[3];  // [1024,1024]
  const float* out_b  = (const float*)d_in[4];  // [1024]
  float* out = (float*)d_out;                   // [8,1024,1024]

  const size_t XN  = 8ull * 1024 * 1024;        // 8388608 tokens*E
  const size_t WQN = 3ull * 1024 * 1024;        // 3145728
  const size_t WON = 1ull * 1024 * 1024;        // 1048576

  half_t* ws    = (half_t*)d_ws;
  half_t* xh    = ws;                 // x as f16            (16 MB)
  half_t* wqkvh = xh + XN;            // qkv_w as f16        ( 6 MB)
  half_t* wouth = wqkvh + WQN;        // out_w as f16        ( 2 MB)
  half_t* qh    = wouth + WON;        // q [b,h,n,d]         (16 MB)
  half_t* kh    = qh + XN;            // k [b,h,n,d]         (16 MB)
  half_t* vth   = kh + XN;            // v^T [b,h,d,n]       (16 MB)
  half_t* attnh = xh;                 // reuse xh region after it is consumed

  // 1) f32 -> f16 conversions
  cvt_f32_f16_kernel<<<dim3((unsigned)(XN / 4 / 256)), dim3(256), 0, stream>>>(x, xh, (int)XN);
  cvt_f32_f16_kernel<<<dim3((unsigned)(WQN / 4 / 256)), dim3(256), 0, stream>>>(qkv_w, wqkvh, (int)WQN);
  cvt_f32_f16_kernel<<<dim3((unsigned)(WON / 4 / 256)), dim3(256), 0, stream>>>(out_w, wouth, (int)WON);

  // 2) QKV projection (8192 x 3072, K=1024)
  qkv_gemm_kernel<<<dim3(24, 64), dim3(128), 0, stream>>>(xh, wqkvh, qkv_b, qh, kh, vth);

  // 3) Flash attention: 8 batches * 16 heads * 64 query tiles
  attn_kernel<<<dim3(8192), dim3(32), 0, stream>>>(qh, kh, vth, attnh);

  // 4) Output projection (8192 x 1024, K=1024)
  out_gemm_kernel<<<dim3(8, 64), dim3(128), 0, stream>>>(attnh, wouth, out_b, out);
}
